// EncoderRNN_59708635349617
// MI455X (gfx1250) — matrix operand order repaired and checked by reading, not compiled, not run on hardware
//
#include <hip/hip_runtime.h>
#include <stdint.h>

#define H 1024
#define G 4096   // 4*H
#define S 512

typedef __attribute__((ext_vector_type(16))) __bf16 v16bf;
typedef __attribute__((ext_vector_type(8)))  float  v8f;

static __device__ __forceinline__ uint16_t f2bf(float f) {
  uint32_t u = __float_as_uint(f);
  uint32_t r = u + 0x7FFFu + ((u >> 16) & 1u);   // round-to-nearest-even
  return (uint16_t)(r >> 16);
}

// ---------------------------------------------------------------------------
// Weight conversion kernels
// ---------------------------------------------------------------------------

// in: [rows, cols] fp32 row-major -> out: [cols, rows] bf16, LDS-tiled 32x32
// so both the reads and the 2-byte writes are coalesced.
__global__ void __launch_bounds__(256)
transpose_to_bf16(const float* __restrict__ in, uint16_t* __restrict__ out,
                  int rows, int cols) {
  __shared__ uint16_t tile[32][33];            // +1 pad kills bank conflicts
  const int c0 = blockIdx.x * 32;              // input col base
  const int r0 = blockIdx.y * 32;              // input row base
  const int tx = threadIdx.x;                  // 0..31
  const int ty = threadIdx.y;                  // 0..7
  for (int i = ty; i < 32; i += 8)
    tile[i][tx] = f2bf(in[(size_t)(r0 + i) * cols + c0 + tx]);
  __syncthreads();
  for (int i = ty; i < 32; i += 8)
    out[(size_t)(c0 + i) * rows + r0 + tx] = tile[tx][i];
}

__global__ void cast_bf16(const float* __restrict__ in,
                          uint16_t* __restrict__ out, int n) {
  int idx = blockIdx.x * 256 + threadIdx.x;
  if (idx < n) out[idx] = f2bf(in[idx]);
}

// Embedding gather -> bf16 sequence [S, H]
__global__ void embed_gather(const int* __restrict__ tokens,
                             const float* __restrict__ emb,
                             uint16_t* __restrict__ xs) {
  int t = blockIdx.x;
  const float* row = emb + (size_t)tokens[t] * H;
  for (int h = threadIdx.x; h < H; h += 256)
    xs[(size_t)t * H + h] = f2bf(row[h]);
}

// Initialize h/c state buffers for both directions from h0/c0
__global__ void init_state(const float* __restrict__ h0,
                           const float* __restrict__ c0,
                           float* __restrict__ hbufs,   // [2 dir][2 parity][H]
                           float* __restrict__ cbufs) { // [2 dir][H]
  int idx = blockIdx.x * 256 + threadIdx.x;
  if (idx >= 2 * H) return;
  int dir = idx / H, j = idx - dir * H;
  hbufs[(size_t)dir * 2 * H + j] = h0[j];     // parity 0 slot
  cbufs[(size_t)dir * H + j]     = c0[j];
}

// ---------------------------------------------------------------------------
// Input projection: gx[s,g] = sum_k xs[s,k] * WxT[k,g] + bx[g] + bh[g]
// M=512, N=4096, K=1024 per direction.  One wave -> 16x64 tile, bf16 WMMA.
// ---------------------------------------------------------------------------
__global__ void __launch_bounds__(128)
input_proj_wmma(const uint16_t* __restrict__ wxt_f,
                const uint16_t* __restrict__ wxt_b,
                const uint16_t* __restrict__ xs,
                const float* __restrict__ bx_f, const float* __restrict__ bh_f,
                const float* __restrict__ bx_b, const float* __restrict__ bh_b,
                float* __restrict__ gx_f, float* __restrict__ gx_b) {
  const int dir = blockIdx.z;
  const uint16_t* wxt = dir ? wxt_b : wxt_f;
  const float*    bx  = dir ? bx_b  : bx_f;
  const float*    bh  = dir ? bh_b  : bh_f;
  float*          gx  = dir ? gx_b  : gx_f;

  const int lane = threadIdx.x & 31;
  const int wave = threadIdx.x >> 5;
  const int n0   = (blockIdx.x * 4 + wave) * 64;  // 64 output cols per wave
  const int s0   = blockIdx.y * 16;               // 16 output rows

  // A layout (16-bit A 16x32): lane = M; lanes 16-31 carry K offset by 8.
  const int m  = lane & 15;
  const int k0 = (lane >> 4) * 8;

  v8f acc0 = {}, acc1 = {}, acc2 = {}, acc3 = {};

  for (int kk = 0; kk < H; kk += 32) {
    union { uint4 q[2]; v16bf v; } A;
    const uint4* ap =
        reinterpret_cast<const uint4*>(xs + (size_t)(s0 + m) * H + kk + k0);
    A.q[0] = ap[0];          // K = k0 .. k0+7
    A.q[1] = ap[2];          // K = k0+16 .. k0+23   (+16 halves = +2 uint4)

    // B layout (32x16, K x N): lane = N (lane&15), K in same order as A.
    const int bn = n0 + (lane & 15);
    union { uint16_t u[16]; v16bf v; } B0, B1, B2, B3;
#pragma unroll
    for (int s = 0; s < 16; ++s) {
      const int kb = kk + k0 + (s & 7) + (s >> 3) * 16;
      const uint16_t* bp = wxt + (size_t)kb * G + bn;
      B0.u[s] = bp[0]; B1.u[s] = bp[16]; B2.u[s] = bp[32]; B3.u[s] = bp[48];
    }

    acc0 = __builtin_amdgcn_wmma_f32_16x16x32_bf16(false, A.v, false, B0.v,
                                                   (short)0, acc0, false, false);
    acc1 = __builtin_amdgcn_wmma_f32_16x16x32_bf16(false, A.v, false, B1.v,
                                                   (short)0, acc1, false, false);
    acc2 = __builtin_amdgcn_wmma_f32_16x16x32_bf16(false, A.v, false, B2.v,
                                                   (short)0, acc2, false, false);
    acc3 = __builtin_amdgcn_wmma_f32_16x16x32_bf16(false, A.v, false, B3.v,
                                                   (short)0, acc3, false, false);
  }

  // C/D layout: VGPR r -> M = r + 8*(lane>=16), N = lane&15.
  const int mrow = s0 + (lane >> 4) * 8;
  const int ncol = n0 + (lane & 15);
#pragma unroll
  for (int r = 0; r < 8; ++r) {
    size_t row = (size_t)(mrow + r) * G;
    gx[row + ncol +  0] = acc0[r] + bx[ncol +  0] + bh[ncol +  0];
    gx[row + ncol + 16] = acc1[r] + bx[ncol + 16] + bh[ncol + 16];
    gx[row + ncol + 32] = acc2[r] + bx[ncol + 32] + bh[ncol + 32];
    gx[row + ncol + 48] = acc3[r] + bx[ncol + 48] + bh[ncol + 48];
  }
}

// ---------------------------------------------------------------------------
// One LSTM time step, both directions in one launch.
// grid = (H/8, 2), block = 256 (8 waves).  One wave per hidden unit j: each
// of the 4 gate rows (1024 bf16, L2-resident) is split 32 ways across the
// wave (32-elem chunk / 64 B per lane -> 2 KB fully-coalesced per row), then
// reduced with a 5-level shfl_xor tree.  2048 waves per step instead of 64,
// and FMA dependency chains of 32 instead of 4096.
// ---------------------------------------------------------------------------
__global__ void __launch_bounds__(256)
lstm_step(const uint16_t* __restrict__ wh_f, const uint16_t* __restrict__ wh_b,
          const float* __restrict__ gx_f, const float* __restrict__ gx_b,
          float* __restrict__ hbufs,   // [2 dir][2 parity][H]
          float* __restrict__ cbufs,   // [2 dir][H]
          float* __restrict__ out,     // [S, 2H]
          int t) {
  const int dir  = blockIdx.y;
  const int wave = threadIdx.x >> 5;
  const int lane = threadIdx.x & 31;
  const int j    = blockIdx.x * 8 + wave;       // hidden unit owned by wave

  const uint16_t* wh = dir ? wh_b : wh_f;
  const float* gx = dir ? (gx_b + (size_t)(S - 1 - t) * G)
                        : (gx_f + (size_t)t * G);
  float* hprev = hbufs + (size_t)dir * 2 * H + (size_t)(t & 1) * H;
  float* hnext = hbufs + (size_t)dir * 2 * H + (size_t)((t & 1) ^ 1) * H;
  float* cbuf  = cbufs + (size_t)dir * H;

  __shared__ float sh[H];
  for (int k = threadIdx.x; k < H; k += 256) sh[k] = hprev[k];
  __syncthreads();

  const float* s = sh + lane * 32;              // this lane's h chunk
  float part[4];
#pragma unroll
  for (int r = 0; r < 4; ++r) {                 // gate order: i, f, o, g
    const int row = r * H + j;
    const uint4* wp =
        reinterpret_cast<const uint4*>(wh + (size_t)row * H) + lane * 4;
    float a0 = 0.0f, a1 = 0.0f;
#pragma unroll
    for (int q = 0; q < 4; ++q) {               // 4 x uint4 = 32 bf16
      uint4 w = wp[q];
      const float* sq = s + q * 8;
      a0 = fmaf(__uint_as_float(w.x << 16),         sq[0], a0);
      a1 = fmaf(__uint_as_float(w.x & 0xFFFF0000u), sq[1], a1);
      a0 = fmaf(__uint_as_float(w.y << 16),         sq[2], a0);
      a1 = fmaf(__uint_as_float(w.y & 0xFFFF0000u), sq[3], a1);
      a0 = fmaf(__uint_as_float(w.z << 16),         sq[4], a0);
      a1 = fmaf(__uint_as_float(w.z & 0xFFFF0000u), sq[5], a1);
      a0 = fmaf(__uint_as_float(w.w << 16),         sq[6], a0);
      a1 = fmaf(__uint_as_float(w.w & 0xFFFF0000u), sq[7], a1);
    }
    part[r] = a0 + a1;
  }

  // Wave-level tree reduction (wave32)
#pragma unroll
  for (int r = 0; r < 4; ++r)
#pragma unroll
    for (int off = 16; off > 0; off >>= 1)
      part[r] += __shfl_xor(part[r], off, 32);

  if (lane == 0) {
    float ig = 1.0f / (1.0f + __expf(-(part[0] + gx[0 * H + j])));
    float fg = 1.0f / (1.0f + __expf(-(part[1] + gx[1 * H + j])));
    float og = 1.0f / (1.0f + __expf(-(part[2] + gx[2 * H + j])));
    float e2 = __expf(2.0f * (part[3] + gx[3 * H + j]));
    float gg = (e2 - 1.0f) / (e2 + 1.0f);       // tanh
    float cn = fg * cbuf[j] + ig * gg;
    cbuf[j] = cn;
    float ec = __expf(2.0f * cn);
    float hn = og * (ec - 1.0f) / (ec + 1.0f);
    hnext[j] = hn;
    int sout = dir ? (S - 1 - t) : t;
    out[(size_t)sout * (2 * H) + dir * H + j] = hn;
  }
}

// ---------------------------------------------------------------------------
// Host launcher
// ---------------------------------------------------------------------------
extern "C" void kernel_launch(void* const* d_in, const int* in_sizes, int n_in,
                              void* d_out, int out_size, void* d_ws,
                              size_t ws_size, hipStream_t stream) {
  (void)in_sizes; (void)n_in; (void)out_size; (void)ws_size;

  const int*   tokens = (const int*)  d_in[0];
  const float* h0     = (const float*)d_in[1];
  const float* c0     = (const float*)d_in[2];
  const float* emb    = (const float*)d_in[3];
  const float* Wx_f   = (const float*)d_in[4];
  const float* Wh_f   = (const float*)d_in[5];
  const float* bx_f   = (const float*)d_in[6];
  const float* bh_f   = (const float*)d_in[7];
  const float* Wx_b   = (const float*)d_in[8];
  const float* Wh_b   = (const float*)d_in[9];
  const float* bx_b   = (const float*)d_in[10];
  const float* bh_b   = (const float*)d_in[11];
  float* out = (float*)d_out;

  // Workspace layout (bytes)
  char* ws = (char*)d_ws;
  const size_t MB = 1u << 20;
  uint16_t* wxt_f = (uint16_t*)(ws +  0 * MB);   // [H, G] bf16 transposed, 8 MB
  uint16_t* wxt_b = (uint16_t*)(ws +  8 * MB);   // 8 MB
  uint16_t* whb_f = (uint16_t*)(ws + 16 * MB);   // [G, H] bf16 row-major, 8 MB
  uint16_t* whb_b = (uint16_t*)(ws + 24 * MB);   // 8 MB
  uint16_t* xs    = (uint16_t*)(ws + 32 * MB);   // [S, H] bf16, 1 MB
  float*    gx_f  = (float*)   (ws + 33 * MB);   // [S, G] f32, 8 MB
  float*    gx_b  = (float*)   (ws + 41 * MB);   // 8 MB
  float*    hbufs = (float*)   (ws + 49 * MB);   // 2*2*H floats
  float*    cbufs = hbufs + 2 * 2 * H;           // 2*H floats

  // 1) Convert weights (Wx transposed for WMMA B-tiles; Wh row-major bf16)
  {
    dim3 tgrid(H / 32, G / 32);                  // (cols/32, rows/32)
    dim3 tblk(32, 8);
    transpose_to_bf16<<<tgrid, tblk, 0, stream>>>(Wx_f, wxt_f, G, H);
    transpose_to_bf16<<<tgrid, tblk, 0, stream>>>(Wx_b, wxt_b, G, H);
    int n = G * H;
    cast_bf16<<<(n + 255) / 256, 256, 0, stream>>>(Wh_f, whb_f, n);
    cast_bf16<<<(n + 255) / 256, 256, 0, stream>>>(Wh_b, whb_b, n);
  }

  // 2) Embedding gather + state init
  embed_gather<<<S, 256, 0, stream>>>(tokens, emb, xs);
  init_state<<<(2 * H + 255) / 256, 256, 0, stream>>>(h0, c0, hbufs, cbufs);

  // 3) Input projection GEMM (WMMA bf16, f32 accumulate), both directions
  {
    dim3 grid(G / 256, S / 16, 2);   // 16 x 32 x 2 blocks, 4 waves each
    input_proj_wmma<<<grid, 128, 0, stream>>>(wxt_f, wxt_b, xs,
                                              bx_f, bh_f, bx_b, bh_b,
                                              gx_f, gx_b);
  }

  // 4) Sequential recurrence: one launch per step, fwd+bwd concurrent.
  {
    dim3 grid(H / 8, 2);             // one wave per hidden unit
    for (int t = 0; t < S; ++t)
      lstm_step<<<grid, 256, 0, stream>>>(whb_f, whb_b, gx_f, gx_b,
                                          hbufs, cbufs, out, t);
  }
}
